// MHN_70643622085021
// MI455X (gfx1250) — matrix operand order repaired and checked by reading, not compiled, not run on hardware
//
#include <hip/hip_runtime.h>

// ================= CDNA5 (gfx1250) wave32 WMMA kernels =================
// All GEMMs: C = alpha*(A[MxK] @ Bm[NxK]^T) + bias, K-contiguous operands.

typedef __attribute__((ext_vector_type(16))) _Float16 v16h;
typedef __attribute__((ext_vector_type(8)))  _Float16 v8h;
typedef __attribute__((ext_vector_type(8)))  float    v8f;
typedef __attribute__((ext_vector_type(4)))  float    v4f;

#define LDS_STRIDE 40  // halves: 32 + 8 pad -> 80B pitch, conflict-free frag loads

// ---- CDNA5 async global->LDS path (ASYNCcnt), with VGPR fallback ----
#if defined(__AMDGCN__) &&                                           \
    __has_builtin(__builtin_amdgcn_global_load_async_to_lds_b128) && \
    __has_builtin(__builtin_amdgcn_s_wait_asynccnt)
#define USE_ASYNC_LDS 1
typedef int v4i_vec __attribute__((vector_size(16)));
typedef __attribute__((address_space(1))) v4i_vec glob_v4i;  // global (AS1)
typedef __attribute__((address_space(3))) v4i_vec lds_v4i;   // LDS    (AS3)
#else
#define USE_ASYNC_LDS 0
#endif

// stage 16 contiguous f32 K-elements into LDS as f16 (zero-fill invalid rows)
__device__ __forceinline__ void stage16(const float* __restrict__ src, bool valid,
                                        _Float16* __restrict__ dst) {
  v8h o0 = {}, o1 = {};
  if (valid) {
    const v4f* q = (const v4f*)src;
    v4f x0 = q[0], x1 = q[1], x2 = q[2], x3 = q[3];
#pragma unroll
    for (int i = 0; i < 4; ++i) {
      o0[i]     = (_Float16)x0[i];
      o0[4 + i] = (_Float16)x1[i];
      o1[i]     = (_Float16)x2[i];
      o1[4 + i] = (_Float16)x3[i];
    }
  }
  *(v8h*)dst       = o0;
  *((v8h*)dst + 1) = o1;
}

// copy 16 contiguous f16 elements global->LDS (async on CDNA5)
__device__ __forceinline__ void copy16_f16(const _Float16* __restrict__ gsrc,
                                           _Float16* __restrict__ ldst) {
#if USE_ASYNC_LDS
  __builtin_amdgcn_global_load_async_to_lds_b128((glob_v4i*)gsrc, (lds_v4i*)ldst, 0, 0);
  __builtin_amdgcn_global_load_async_to_lds_b128((glob_v4i*)(gsrc + 8),
                                                 (lds_v4i*)(ldst + 8), 0, 0);
#else
  const v8h* q = (const v8h*)gsrc;
  *(v8h*)ldst       = q[0];
  *((v8h*)ldst + 1) = q[1];
#endif
}

__device__ __forceinline__ void async_wait_all() {
#if USE_ASYNC_LDS
  __builtin_amdgcn_s_wait_asynccnt(0);
#endif
}

// fragment loaders per ISA 7.12.2 (wave32, 16-bit A/B layouts)
__device__ __forceinline__ v16h load_afrag(const _Float16* s, int akoff) {
  v8h lo = *(const v8h*)(s + akoff);
  v8h hi = *(const v8h*)(s + akoff + 16);
  return __builtin_shufflevector(lo, hi, 0, 1, 2, 3, 4, 5, 6, 7,
                                 8, 9, 10, 11, 12, 13, 14, 15);
}
__device__ __forceinline__ v16h load_bfrag(const _Float16* s, int bkoff) {
  v8h lo = *(const v8h*)(s + bkoff);
  v8h hi = *(const v8h*)(s + bkoff + 16);
  return __builtin_shufflevector(lo, hi, 0, 1, 2, 3, 4, 5, 6, 7,
                                 8, 9, 10, 11, 12, 13, 14, 15);
}

// =====================================================================
// 1) f32 -> f16 weight conversion (one-shot, 12 MB traffic)
// =====================================================================
__global__ void __launch_bounds__(256)
cvt_f32_f16(const float* __restrict__ src, _Float16* __restrict__ dst, size_t n) {
  size_t i = ((size_t)blockIdx.x * 256 + threadIdx.x) * 8;
  if (i + 8 <= n) {
    v4f a = *(const v4f*)(src + i);
    v4f b = *(const v4f*)(src + i + 4);
    v8h o;
#pragma unroll
    for (int j = 0; j < 4; ++j) {
      o[j]     = (_Float16)a[j];
      o[4 + j] = (_Float16)b[j];
    }
    *(v8h*)(dst + i) = o;
  }
}

// =====================================================================
// 2) N-resident encoder GEMM: C[M x 1024] = A[M x K] @ Wh[1024 x K]^T + bias
//    Block = 32 rows x all 1024 cols -> A (templates) read from HBM ONCE.
//    64KB B-tile in CDNA5's 320KB WGP LDS; staged with async-to-LDS.
// =====================================================================
#define ENC_BM 32
#define ENC_BN 1024
#define ENC_BK 32

__global__ void __launch_bounds__(256)
encoder_gemm(const float* __restrict__ A, const _Float16* __restrict__ Wh,
             const float* __restrict__ bias, _Float16* __restrict__ C,
             int M, int K) {
  __shared__ alignas(16) _Float16 As[ENC_BM * LDS_STRIDE];   //  2.5 KB
  __shared__ alignas(16) _Float16 Bs[ENC_BN * LDS_STRIDE];   // 80   KB

  const int tid  = threadIdx.x;
  const int lane = tid & 31;
  const int wave = tid >> 5;     // 8 waves
  const int wn   = wave * 128;   // each wave owns 128 of the 1024 columns
  const int m0   = blockIdx.x * ENC_BM;

  // A staging (f32->f16 convert): waves 0-1 stage the 32x32 tile
  const int  arow = tid >> 1;
  const int  aseg = (tid & 1) * 16;
  const bool doA  = (tid < 64);
  const bool va   = doA && (m0 + arow) < M;
  const size_t aoff = (size_t)(m0 + (doA ? arow : 0)) * (size_t)K + aseg;

  // B staging: 2048 (row,seg) units, 8 per thread, pure f16 copy -> async
  const int brow0 = tid >> 1;
  const int bseg  = (tid & 1) * 16;

  const int frow  = lane & 15;
  const int akoff = (lane >> 4) * 8;
  const int bkoff = (lane >> 4) * 8;
  const int crow  = (lane >> 4) * 8;
  const int ccol  = lane & 15;

  v8f acc[2][8] = {};

  for (int k0 = 0; k0 < K; k0 += ENC_BK) {
#pragma unroll
    for (int j = 0; j < 8; ++j) {
      const int row = brow0 + 128 * j;
      copy16_f16(Wh + (size_t)row * (size_t)K + k0 + bseg,
                 &Bs[row * LDS_STRIDE + bseg]);
    }
    if (doA)
      stage16(A + aoff + k0, va, &As[arow * LDS_STRIDE + aseg]);
    if (doA && k0 + ENC_BK < K)
      __builtin_prefetch(A + aoff + k0 + ENC_BK, 0, 0);  // global_prefetch_b8
    async_wait_all();
    __syncthreads();

    v16h afrag[2];
#pragma unroll
    for (int mi = 0; mi < 2; ++mi)
      afrag[mi] = load_afrag(&As[(mi * 16 + frow) * LDS_STRIDE], akoff);
#pragma unroll
    for (int ni = 0; ni < 8; ++ni) {
      v16h bfrag = load_bfrag(&Bs[(wn + ni * 16 + frow) * LDS_STRIDE], bkoff);
#pragma unroll
      for (int mi = 0; mi < 2; ++mi)
        acc[mi][ni] = __builtin_amdgcn_wmma_f32_16x16x32_f16(
            false, afrag[mi], false, bfrag, (short)0, acc[mi][ni], false, false);
    }
    __syncthreads();
  }

#pragma unroll
  for (int ni = 0; ni < 8; ++ni) {
    const int   gc = wn + ni * 16 + ccol;  // always < 1024
    const float bv = bias[gc];
#pragma unroll
    for (int mi = 0; mi < 2; ++mi) {
      const int gr0 = m0 + mi * 16 + crow;
#pragma unroll
      for (int v = 0; v < 8; ++v) {
        const int gr = gr0 + v;
        if (gr < M) C[(size_t)gr * ENC_BN + gc] = (_Float16)(acc[mi][ni][v] + bv);
      }
    }
  }
}

// =====================================================================
// 3) out-GEMM: C[M x N] = alpha * A[M x K] @ Bm[N x K]^T   (f16 in, f32 out)
//    256x128 tiles -> X streamed from HBM only twice (M=512).
//    Edge rows handled by address clamping (garbage feeds unstored cols).
// =====================================================================
#define G3_BM 256
#define G3_BN 128
#define G3_BK 32

__global__ void __launch_bounds__(256)
gemm_f16_abT(const _Float16* __restrict__ A, const _Float16* __restrict__ Bm,
             float* __restrict__ C, int M, int N, int K, float alpha) {
  __shared__ alignas(16) _Float16 As[G3_BM * LDS_STRIDE];  // 20 KB
  __shared__ alignas(16) _Float16 Bs[G3_BN * LDS_STRIDE];  // 10 KB

  const int tid  = threadIdx.x;
  const int lane = tid & 31;
  const int wave = tid >> 5;
  const int wm   = (wave & 1) * 128;   // 2x4 wave grid, 128x32 per wave
  const int wn   = (wave >> 1) * 32;

  const int m0 = blockIdx.y * G3_BM;
  const int n0 = blockIdx.x * G3_BN;

  // staging units: A has 512 (row,seg) units -> 2/thread; B has 256 -> 1/thread
  const int srow = tid >> 1;
  const int sseg = (tid & 1) * 16;
  int ar0 = m0 + srow;        if (ar0 >= M) ar0 = 0;   // clamp (always-load)
  int ar1 = m0 + srow + 128;  if (ar1 >= M) ar1 = 0;
  int br  = n0 + srow;        if (br  >= N) br  = 0;
  const size_t aoff0 = (size_t)ar0 * (size_t)K + sseg;
  const size_t aoff1 = (size_t)ar1 * (size_t)K + sseg;
  const size_t boff  = (size_t)br  * (size_t)K + sseg;

  const int frow  = lane & 15;
  const int akoff = (lane >> 4) * 8;
  const int bkoff = (lane >> 4) * 8;

  v8f acc[8][2] = {};

  for (int k0 = 0; k0 < K; k0 += G3_BK) {
    copy16_f16(A + aoff0 + k0, &As[srow * LDS_STRIDE + sseg]);
    copy16_f16(A + aoff1 + k0, &As[(srow + 128) * LDS_STRIDE + sseg]);
    copy16_f16(Bm + boff + k0, &Bs[srow * LDS_STRIDE + sseg]);
    async_wait_all();
    __syncthreads();

    v16h bfrag[2];
#pragma unroll
    for (int ni = 0; ni < 2; ++ni)
      bfrag[ni] = load_bfrag(&Bs[(wn + ni * 16 + frow) * LDS_STRIDE], bkoff);
#pragma unroll
    for (int mi = 0; mi < 8; ++mi) {
      v16h afrag = load_afrag(&As[(wm + mi * 16 + frow) * LDS_STRIDE], akoff);
#pragma unroll
      for (int ni = 0; ni < 2; ++ni)
        acc[mi][ni] = __builtin_amdgcn_wmma_f32_16x16x32_f16(
            false, afrag, false, bfrag[ni], (short)0, acc[mi][ni], false, false);
    }
    __syncthreads();
  }

  const int crow = (lane >> 4) * 8;
  const int ccol = lane & 15;
#pragma unroll
  for (int ni = 0; ni < 2; ++ni) {
    const int gc = n0 + wn + ni * 16 + ccol;
    if (gc >= N) continue;
#pragma unroll
    for (int mi = 0; mi < 8; ++mi) {
      const int gr0 = m0 + wm + mi * 16 + crow;
#pragma unroll
      for (int v = 0; v < 8; ++v) {
        const int gr = gr0 + v;
        if (gr < M) C[(size_t)gr * (size_t)N + gc] = alpha * acc[mi][ni][v];
      }
    }
  }
}

// =====================================================================
extern "C" void kernel_launch(void* const* d_in, const int* in_sizes, int n_in,
                              void* d_out, int out_size, void* d_ws, size_t ws_size,
                              hipStream_t stream) {
  const float* m      = (const float*)d_in[0];  // [B, FP]
  const float* templ  = (const float*)d_in[1];  // [T, FP]
  const float* W_mol  = (const float*)d_in[2];  // [A, FP]
  const float* b_mol  = (const float*)d_in[3];  // [A]
  const float* W_temp = (const float*)d_in[4];  // [A, FP]
  const float* b_temp = (const float*)d_in[5];  // [A]
  float* out = (float*)d_out;

  const int nFP = 2048;
  const int nA  = in_sizes[3];        // 1024
  const int nB  = in_sizes[0] / nFP;  // 512
  const int nT  = in_sizes[1] / nFP;  // 50000
  const size_t nW = (size_t)nA * nFP; // 2M elems per weight matrix

  // workspace layout (f16): Xi | X | Wmol_h | Wtemp_h  (~111.5 MB)
  _Float16* Xi      = (_Float16*)d_ws;
  _Float16* X       = Xi + (size_t)nB * nA;
  _Float16* Wmol_h  = X + (size_t)nT * nA;
  _Float16* Wtemp_h = Wmol_h + nW;

  const dim3 blk(256);

  // 0) one-shot weight conversion (weights then live hot in L2)
  cvt_f32_f16<<<dim3((unsigned)(nW / 8 / 256)), blk, 0, stream>>>(W_mol, Wmol_h, nW);
  cvt_f32_f16<<<dim3((unsigned)(nW / 8 / 256)), blk, 0, stream>>>(W_temp, Wtemp_h, nW);

  // 1) Xi = m @ W_mol^T + b_mol
  encoder_gemm<<<dim3((nB + ENC_BM - 1) / ENC_BM), blk, 0, stream>>>(
      m, Wmol_h, b_mol, Xi, nB, nFP);

  // 2) X = templates @ W_temp^T + b_temp   (templates read once)
  encoder_gemm<<<dim3((nT + ENC_BM - 1) / ENC_BM), blk, 0, stream>>>(
      templ, Wtemp_h, b_temp, X, nT, nFP);

  // 3) out = 0.125 * Xi @ X^T
  gemm_f16_abT<<<dim3((nT + G3_BN - 1) / G3_BN, (nB + G3_BM - 1) / G3_BM), blk, 0, stream>>>(
      Xi, X, out, nB, nT, nA, 0.125f);
}
